// PatchGCN_Surv_49203145343049
// MI455X (gfx1250) — compile-verified, matrix operand order checked by reading, not run on hardware
//
#include <hip/hip_runtime.h>
#include <hip/hip_bf16.h>
#include <cstdint>

// ---------------------------------------------------------------------------
// PatchGCN survival model on MI455X (gfx1250, wave32, WMMA).
//
// GEMMs use v_wmma_f32_16x16x32_f16 (f16 in, f32 acc). Weights are pre-packed
// once per launch into the exact per-lane WMMA B-fragment layout (f16), so the
// inner loop is: 1x global_load_b128 (A tile stage) + ds_store_b64 +
// 2x ds_load_b128 (A frag) + 4x global_load_b128 (two B frags) + 2x wmma.
// 50000 rows = 3125 * 16-row tiles exactly; K/N dims are multiples of 32/64,
// so EXEC is all-ones at every WMMA (ISA requirement), no tails anywhere.
// Edge scatter-softmax (~150M f32 atomics, ~600MB traffic vs 23.3TB/s HBM)
// dominates runtime and uses CAS-based float max + global_atomic_add_f32.
// ---------------------------------------------------------------------------

#define N_NODES 50000
#define N_EDGES 400000
#define FEAT    384
#define HID     128
#define DIM     512
#define EPSG    1e-7f

typedef __attribute__((ext_vector_type(16))) _Float16 v16h;
typedef __attribute__((ext_vector_type(4)))  _Float16 v4h;
typedef __attribute__((ext_vector_type(8)))  float    v8f;

// ---------------- float atomic max via CAS --------------------------------
__device__ __forceinline__ void atomicMaxF(float* addr, float val) {
  int old = __float_as_int(*addr);
  while (__int_as_float(old) < val) {
    int assumed = old;
    old = atomicCAS((int*)addr, assumed, __float_as_int(val));
    if (old == assumed) break;
  }
}

// ---------------- weight pre-pack: f32 [K,Nc] -> f16 B fragments ----------
// Layout: frag idx = (ntile*(K/32) + kblock)*32 + lane, 16 halves each.
// Lane(n = lane&15, hi = lane>>4), element pair j: K = ((j<4)?0:16)+8*hi+2*(j&3).
__global__ void pack_w_kernel(const float* __restrict__ W,
                              _Float16* __restrict__ Wp, int K, int Nc) {
  const int idx = blockIdx.x * blockDim.x + threadIdx.x;
  const int total = (Nc / 16) * (K / 32) * 32;
  if (idx >= total) return;
  const int lane = idx & 31;
  const int kb   = (idx >> 5) % (K / 32);
  const int nt   = (idx >> 5) / (K / 32);
  const int n = lane & 15, hi = lane >> 4;
  const int col = nt * 16 + n, k0 = kb * 32;
  _Float16* dp = Wp + (size_t)idx * 16;
#pragma unroll
  for (int j = 0; j < 8; ++j) {
    int base = ((j < 4) ? 0 : 16) + 8 * hi + 2 * (j & 3);
    dp[2 * j]     = (_Float16)W[(size_t)(k0 + base) * Nc + col];
    dp[2 * j + 1] = (_Float16)W[(size_t)(k0 + base + 1) * Nc + col];
  }
}

// ---------------- WMMA GEMM: C = act(A @ W + bias) ------------------------
// A: [M,K] f32 (row stride lda, col offset aoff); Wp: packed f16 fragments.
// 4 waves/WG; each wave owns TWO 16x16 N-tiles sharing one A fragment
// (2 wmma per K-step). Block covers 16 rows x 128 cols.
// ACT: 0=none 1=relu 2=tanh 3=sigmoid. K compile-time -> fully unrolled.
template <int ACT, int K>
__global__ void wmma_gemm_kernel(const float* __restrict__ A, int lda, int aoff,
                                 const _Float16* __restrict__ Wp,
                                 const float* __restrict__ bias,
                                 float* __restrict__ C, int ldc, int coff,
                                 int Nc) {
  __shared__ _Float16 As[16 * 32];   // one 16x32 A tile per K-step (1 KB)
  const int tid  = threadIdx.x;      // 0..127 (4 waves)
  const int wave = tid >> 5;
  const int lane = tid & 31;
  const int row0 = blockIdx.x * 16;
  const int nt0  = (blockIdx.y * 4 + wave) * 2;   // first of two 16-col tiles
  const int m  = lane & 15;
  const int hi = lane >> 4;
  constexpr int KB = K / 32;

  v8f acc0 = {0.f, 0.f, 0.f, 0.f, 0.f, 0.f, 0.f, 0.f};
  v8f acc1 = {0.f, 0.f, 0.f, 0.f, 0.f, 0.f, 0.f, 0.f};

#pragma unroll
  for (int kb = 0; kb < KB; ++kb) {
    __syncthreads();                 // protect previous iteration's reads
    // Stage A tile: one float4 per thread (coalesced b128), pack to f16.
    {
      const int e0 = tid * 4, r = e0 >> 5, c = e0 & 31;
      float4 a4 = *(const float4*)(A + (size_t)(row0 + r) * lda + aoff + kb * 32 + c);
      v4h h4 = {(_Float16)a4.x, (_Float16)a4.y, (_Float16)a4.z, (_Float16)a4.w};
      *(v4h*)(&As[e0]) = h4;         // ds_store_b64
    }
    __syncthreads();

    // A fragment from LDS (ISA 7.12.2 16-bit A 16x32 layout).
    v16h af;
#pragma unroll
    for (int j = 0; j < 8; ++j) {
      int base = ((j < 4) ? 0 : 16) + 8 * hi + 2 * (j & 3);
      af[2 * j]     = As[m * 32 + base];
      af[2 * j + 1] = As[m * 32 + base + 1];
    }
    // B fragments: contiguous 32B per lane, coalesced across the wave.
    const v16h bf0 = *(const v16h*)(Wp + ((size_t)(nt0 * KB + kb) * 32 + lane) * 16);
    const v16h bf1 = *(const v16h*)(Wp + ((size_t)((nt0 + 1) * KB + kb) * 32 + lane) * 16);
    if (kb + 1 < KB)                 // global_prefetch_b8 of next B fragment
      __builtin_prefetch(Wp + ((size_t)(nt0 * KB + kb + 1) * 32 + lane) * 16, 0, 1);

    acc0 = __builtin_amdgcn_wmma_f32_16x16x32_f16(
        false, af, false, bf0, (short)0, acc0, false, false);
    acc1 = __builtin_amdgcn_wmma_f32_16x16x32_f16(
        false, af, false, bf1, (short)0, acc1, false, false);
  }

  // Epilogue: lane holds C[row0 + r + 8*hi][nt*16 + m] in acc[r].
  const int n0 = nt0 * 16 + m, n1 = n0 + 16;
  const float bv0 = bias[n0], bv1 = bias[n1];
#pragma unroll
  for (int r = 0; r < 8; ++r) {
    const size_t rowoff = (size_t)(row0 + r + 8 * hi) * ldc + coff;
    float v0 = acc0[r] + bv0, v1 = acc1[r] + bv1;
    if      (ACT == 1) { v0 = fmaxf(v0, 0.f);            v1 = fmaxf(v1, 0.f); }
    else if (ACT == 2) { v0 = tanhf(v0);                 v1 = tanhf(v1); }
    else if (ACT == 3) { v0 = 1.f / (1.f + __expf(-v0)); v1 = 1.f / (1.f + __expf(-v1)); }
    C[rowoff + n0] = v0;
    C[rowoff + n1] = v1;
  }
}

// ---------------- LayerNorm (wave per row, wave32 shuffles) ---------------
__global__ void ln_kernel(const float* __restrict__ src, int lds_, int soff, int D,
                          const float* __restrict__ g, const float* __restrict__ bt,
                          float* __restrict__ dst, int ldd, int doff,
                          const float* __restrict__ res, int ldr, int roff, int N) {
  const int lane = threadIdx.x & 31;
  const int row  = blockIdx.x * 4 + (threadIdx.x >> 5);
  if (row >= N) return;
  const float* sr = src + (size_t)row * lds_ + soff;
  const int per = D >> 5;            // 4 (D=128) or 8 (D=256)
  float v[8], s = 0.f, s2 = 0.f;
  for (int j = 0; j < per; ++j) {
    float x = sr[lane + j * 32];
    v[j] = x; s += x; s2 += x * x;
  }
  for (int o = 16; o; o >>= 1) { s += __shfl_xor(s, o); s2 += __shfl_xor(s2, o); }
  const float mean = s / D;
  const float inv  = rsqrtf(s2 / D - mean * mean + 1e-5f);
  float* dr = dst + (size_t)row * ldd + doff;
  const float* rr = res ? res + (size_t)row * ldr + roff : nullptr;
  for (int j = 0; j < per; ++j) {
    int c = lane + j * 32;
    float y = fmaxf((v[j] - mean) * inv * g[c] + bt[c], 0.f);
    dr[c] = res ? rr[c] + y : y;
  }
}

// ---------------- edge scatter-softmax (GENConv aggregation) --------------
__global__ void edge_max_kernel(const long long* __restrict__ src,
                                const long long* __restrict__ dst,
                                const float* __restrict__ h, int ldh, int hoff,
                                const float* __restrict__ tptr,
                                float* __restrict__ mbuf, int E) {
  int e = blockIdx.x * blockDim.x + threadIdx.x;
  if (e >= E) return;
  const float tv = tptr[0];
  const long long s = src[e], d = dst[e];
  const float4* hr = (const float4*)(h + (size_t)s * ldh + hoff);
  float* mr = mbuf + (size_t)d * HID;
  for (int j = 0; j < HID / 4; ++j) {
    float4 x = hr[j];
    atomicMaxF(&mr[4 * j + 0], (fmaxf(x.x, 0.f) + EPSG) * tv);
    atomicMaxF(&mr[4 * j + 1], (fmaxf(x.y, 0.f) + EPSG) * tv);
    atomicMaxF(&mr[4 * j + 2], (fmaxf(x.z, 0.f) + EPSG) * tv);
    atomicMaxF(&mr[4 * j + 3], (fmaxf(x.w, 0.f) + EPSG) * tv);
  }
}

__global__ void edge_sum_kernel(const long long* __restrict__ src,
                                const long long* __restrict__ dst,
                                const float* __restrict__ h, int ldh, int hoff,
                                const float* __restrict__ tptr,
                                const float* __restrict__ mbuf,
                                float* __restrict__ den, float* __restrict__ num,
                                int E) {
  int e = blockIdx.x * blockDim.x + threadIdx.x;
  if (e >= E) return;
  const float tv = tptr[0];
  const long long s = src[e], d = dst[e];
  const float4* hr = (const float4*)(h + (size_t)s * ldh + hoff);
  const size_t base = (size_t)d * HID;
  for (int j = 0; j < HID / 4; ++j) {
    float4 x = hr[j];
    float msg[4] = {fmaxf(x.x, 0.f) + EPSG, fmaxf(x.y, 0.f) + EPSG,
                    fmaxf(x.z, 0.f) + EPSG, fmaxf(x.w, 0.f) + EPSG};
#pragma unroll
    for (int q = 0; q < 4; ++q) {
      size_t i = base + 4 * j + q;
      float ev = __expf(msg[q] * tv - mbuf[i]);
      atomicAdd(&den[i], ev);
      atomicAdd(&num[i], ev * msg[q]);
    }
  }
}

__global__ void init_msg_kernel(float* m, float* den, float* num, int n) {
  int i = blockIdx.x * blockDim.x + threadIdx.x;
  if (i < n) { m[i] = -1e30f; den[i] = 0.f; num[i] = 0.f; }
}

__global__ void agg_finalize_kernel(const float* __restrict__ num,
                                    const float* __restrict__ den,
                                    const float* __restrict__ h, int ldh, int hoff,
                                    float* __restrict__ aggout, int n) {
  int i = blockIdx.x * blockDim.x + threadIdx.x;
  if (i >= n) return;
  int node = i >> 7, c = i & (HID - 1);
  float d = den[i];
  float a = d > 0.f ? num[i] / d : 0.f;
  aggout[i] = a + h[(size_t)node * ldh + hoff + c];
}

// ---------------- attention score: A[row] = (tanh·sigm)·c_W + c_b ---------
__global__ void attn_score_kernel(const float* __restrict__ ta,
                                  const float* __restrict__ tb,
                                  const float* __restrict__ cw,
                                  const float* __restrict__ cb,
                                  float* __restrict__ Aout, int N) {
  const int lane = threadIdx.x & 31;
  const int row  = blockIdx.x * 4 + (threadIdx.x >> 5);
  if (row >= N) return;
  const size_t b = (size_t)row * DIM;
  float s = 0.f;
  for (int j = 0; j < DIM / 32; ++j) {
    int i = lane + j * 32;
    s += ta[b + i] * tb[b + i] * cw[i];
  }
  for (int o = 16; o; o >>= 1) s += __shfl_xor(s, o);
  if (lane == 0) Aout[row] = s + cb[0];
}

// ---------------- batch softmax reductions + pooling ----------------------
__global__ void init_small_kernel(float* bmax, float* bden, float* pooled) {
  int i = threadIdx.x;               // 1024 threads
  pooled[i] = 0.f;
  if (i < 2) { bmax[i] = -1e30f; bden[i] = 0.f; }
}

__global__ void batch_max_kernel(const float* A, float* bmax, int N, int Mper) {
  int i = blockIdx.x * blockDim.x + threadIdx.x;
  if (i < N) atomicMaxF(&bmax[i / Mper], A[i]);
}

__global__ void batch_den_kernel(const float* A, const float* bmax, float* bden,
                                 float* outA, int N, int Mper) {
  int i = blockIdx.x * blockDim.x + threadIdx.x;
  if (i >= N) return;
  outA[i] = A[i];                    // A_path output is the *raw* scores
  atomicAdd(&bden[i / Mper], __expf(A[i] - bmax[i / Mper]));
}

__global__ void pooled_kernel(const float* __restrict__ A,
                              const float* __restrict__ bmax,
                              const float* __restrict__ bden,
                              const float* __restrict__ hp,
                              float* __restrict__ pooled, int Mper, int CH) {
  const int b = blockIdx.y, c = threadIdx.x;     // 512 threads
  const int row0 = b * Mper + blockIdx.x * CH;
  const float mx = bmax[b], dn = bden[b];
  float acc = 0.f;
  for (int i = 0; i < CH; ++i) {
    int r = row0 + i;
    acc += (__expf(A[r] - mx) / dn) * hp[(size_t)r * DIM + c];
  }
  atomicAdd(&pooled[b * DIM + c], acc);
}

// ---------------- head: rho -> cls -> hazards/S/argmax --------------------
__global__ void head_kernel(const float* __restrict__ pooled,
                            const float* __restrict__ rhoW, const float* __restrict__ rhob,
                            const float* __restrict__ clsW, const float* __restrict__ clsb,
                            float* __restrict__ out) {
  __shared__ float sp[DIM], sh[DIM], sl[4];
  const int b = blockIdx.x, c = threadIdx.x;     // 512 threads
  sp[c] = pooled[b * DIM + c];
  __syncthreads();
  float s = 0.f;
  for (int k = 0; k < DIM; ++k) s += sp[k] * rhoW[k * DIM + c];
  sh[c] = fmaxf(s + rhob[c], 0.f);
  __syncthreads();
  if (c < 4) {
    float t = 0.f;
    for (int k = 0; k < DIM; ++k) t += sh[k] * clsW[k * 4 + c];
    sl[c] = t + clsb[c];
  }
  __syncthreads();
  if (c == 0) {
    float best = sl[0], S = 1.f; int arg = 0;
    for (int j = 0; j < 4; ++j) {
      if (sl[j] > best) { best = sl[j]; arg = j; }
      float hz = 1.f / (1.f + __expf(-sl[j]));
      out[b * 4 + j] = hz;           // hazards [2,4]
      S *= (1.f - hz);
      out[8 + b * 4 + j] = S;        // S [2,4]
    }
    out[16 + b] = (float)arg;        // Y_hat [2,1]
  }
}

// ---------------- host-side helpers ---------------------------------------
static void pack_w(const float* W, _Float16* Wp, int K, int Nc, hipStream_t st) {
  int total = (Nc / 16) * (K / 32) * 32;
  pack_w_kernel<<<(total + 255) / 256, 256, 0, st>>>(W, Wp, K, Nc);
}

template <int ACT, int K>
static void gemm(const float* A, int lda, int aoff, const _Float16* Wp,
                 const float* bias, float* C, int ldc, int coff,
                 int Nc, hipStream_t st) {
  dim3 g(N_NODES / 16, Nc / 128), blk(128);
  wmma_gemm_kernel<ACT, K><<<g, blk, 0, st>>>(A, lda, aoff, Wp, bias, C, ldc, coff, Nc);
}

extern "C" void kernel_launch(void* const* d_in, const int* in_sizes, int n_in,
                              void* d_out, int out_size, void* d_ws, size_t ws_size,
                              hipStream_t stream) {
  (void)in_sizes; (void)n_in; (void)out_size; (void)ws_size;
  // ---- inputs (setup_inputs() insertion order) ----
  const float*     x  = (const float*)d_in[0];
  const long long* ei = (const long long*)d_in[1];      // int64 [2,E]
  // d_in[2] = batch_size (device scalar; B=2 compile-time per setup)
  const float* fcW  = (const float*)d_in[3];
  const float* fcb  = (const float*)d_in[4];
  // convs: base 5 + 7*i : t, W1, b1, ln_g, ln_b, W2, b2
  // outer_ln: base 26 + 2*j : g, b
  const float* phiW = (const float*)d_in[30];
  const float* phib = (const float*)d_in[31];
  const float* aW   = (const float*)d_in[32];
  const float* ab_  = (const float*)d_in[33];
  const float* bW   = (const float*)d_in[34];
  const float* bb_  = (const float*)d_in[35];
  const float* cW   = (const float*)d_in[36];
  const float* cb_  = (const float*)d_in[37];
  const float* rhoW = (const float*)d_in[38];
  const float* rhob = (const float*)d_in[39];
  const float* clsW = (const float*)d_in[40];
  const float* clsb = (const float*)d_in[41];

  const long long* src = ei;
  const long long* dst = ei + N_EDGES;
  float* out = (float*)d_out;
  const size_t N = N_NODES;

  // ---- workspace layout (floats) ----
  float* ws   = (float*)d_ws;
  float* xcat = ws;                  // [N,512]: cols 0-127 fc | 128-255 L0 | 256-383 L1 | 384-511 L2
  float* bufB = ws + N * 512;        // phase A: h1 | m | den ; phase B: hp[N,512]
  float* bufC = ws + N * 1024;       // phase A: num | agg | t ; phase B: ta[N,512]
  float* bufD = ws + N * 1536;       // phase B: tb[N,512]
  float* Abuf = ws + N * 2048;       // [N]
  float* bmax = Abuf + N;            // [2]
  float* bden = bmax + 2;            // [2]
  float* pooled = bden + 2;          // [2,512]
  _Float16* wp = (_Float16*)(ws + N * 2048 + N + 1040);   // packed f16 weights

  float* h1  = bufB;
  float* mB  = bufB + N * 256;
  float* den = bufB + N * 384;
  float* num = bufC;
  float* agg = bufC + N * 128;
  float* tbf = bufC + N * 256;

  // ---- pack all weights to f16 WMMA B-fragment layout (once / launch) ----
  size_t off = 0;
  _Float16* fcWp = wp + off; off += (size_t)FEAT * HID;
  _Float16 *W1p[3], *W2p[3];
  for (int i = 0; i < 3; ++i) {
    W1p[i] = wp + off; off += (size_t)HID * 2 * HID;
    W2p[i] = wp + off; off += (size_t)2 * HID * HID;
  }
  _Float16* phiWp = wp + off; off += (size_t)DIM * DIM;
  _Float16* aWp   = wp + off; off += (size_t)DIM * DIM;
  _Float16* bWp   = wp + off; off += (size_t)DIM * DIM;

  pack_w(fcW, fcWp, FEAT, HID, stream);
  for (int i = 0; i < 3; ++i) {
    pack_w((const float*)d_in[5 + 7 * i + 1], W1p[i], HID, 2 * HID, stream);
    pack_w((const float*)d_in[5 + 7 * i + 5], W2p[i], 2 * HID, HID, stream);
  }
  pack_w(phiW, phiWp, DIM, DIM, stream);
  pack_w(aW,   aWp,   DIM, DIM, stream);
  pack_w(bW,   bWp,   DIM, DIM, stream);

  // ---- fc: xcat[:,0:128] = relu(x @ fcW + fcb) ----
  gemm<1, FEAT>(x, FEAT, 0, fcWp, fcb, xcat, DIM, 0, HID, stream);

  // ---- 3 GENConv layers ----
  const int nh = N_NODES * HID;
  for (int i = 0; i < 3; ++i) {
    const int base = 5 + 7 * i;
    const float* tP = (const float*)d_in[base + 0];
    const float* b1 = (const float*)d_in[base + 2];
    const float* lg = (const float*)d_in[base + 3];
    const float* lb = (const float*)d_in[base + 4];
    const float* b2 = (const float*)d_in[base + 6];
    const int hoff = i * HID;        // input feature slice of xcat

    init_msg_kernel<<<(nh + 255) / 256, 256, 0, stream>>>(mB, den, num, nh);
    edge_max_kernel<<<(N_EDGES + 255) / 256, 256, 0, stream>>>(
        src, dst, xcat, DIM, hoff, tP, mB, N_EDGES);
    edge_sum_kernel<<<(N_EDGES + 255) / 256, 256, 0, stream>>>(
        src, dst, xcat, DIM, hoff, tP, mB, den, num, N_EDGES);
    agg_finalize_kernel<<<(nh + 255) / 256, 256, 0, stream>>>(
        num, den, xcat, DIM, hoff, agg, nh);

    // MLP: h1 = relu(LN(agg @ W1 + b1)); t = h1 @ W2 + b2
    gemm<0, HID>(agg, HID, 0, W1p[i], b1, h1, 2 * HID, 0, 2 * HID, stream);
    ln_kernel<<<N_NODES / 4, 128, 0, stream>>>(h1, 2 * HID, 0, 2 * HID, lg, lb,
                                               h1, 2 * HID, 0, nullptr, 0, 0, N_NODES);
    if (i == 0) {                    // bare conv: next h = t, straight into xcat
      gemm<0, 2 * HID>(h1, 2 * HID, 0, W2p[i], b2, xcat, DIM, HID, HID, stream);
    } else {                         // res block: h_new = h + relu(LN(t))
      const float* og = (const float*)d_in[26 + 2 * (i - 1)];
      const float* ob = (const float*)d_in[27 + 2 * (i - 1)];
      gemm<0, 2 * HID>(h1, 2 * HID, 0, W2p[i], b2, tbf, HID, 0, HID, stream);
      ln_kernel<<<N_NODES / 4, 128, 0, stream>>>(tbf, HID, 0, HID, og, ob,
                                                 xcat, DIM, (i + 1) * HID,
                                                 xcat, DIM, i * HID, N_NODES);
    }
  }

  // ---- attention head ----
  float* hp = bufB;                  // conv scratch no longer live
  float* ta = bufC;
  float* tb = bufD;
  gemm<1, DIM>(xcat, DIM, 0, phiWp, phib, hp, DIM, 0, DIM, stream);  // relu
  gemm<2, DIM>(hp,   DIM, 0, aWp,   ab_,  ta, DIM, 0, DIM, stream);  // tanh
  gemm<3, DIM>(hp,   DIM, 0, bWp,   bb_,  tb, DIM, 0, DIM, stream);  // sigmoid
  attn_score_kernel<<<N_NODES / 4, 128, 0, stream>>>(ta, tb, cW, cb_, Abuf, N_NODES);

  const int Mper = N_NODES / 2;      // 25000 rows per batch item
  init_small_kernel<<<1, 1024, 0, stream>>>(bmax, bden, pooled);
  batch_max_kernel<<<(N_NODES + 255) / 256, 256, 0, stream>>>(Abuf, bmax, N_NODES, Mper);
  batch_den_kernel<<<(N_NODES + 255) / 256, 256, 0, stream>>>(Abuf, bmax, bden,
                                                              out + 18, N_NODES, Mper);
  const int CH = 500;                // 25000 / 50 chunks
  pooled_kernel<<<dim3(Mper / CH, 2), DIM, 0, stream>>>(Abuf, bmax, bden, hp, pooled, Mper, CH);
  head_kernel<<<2, DIM, 0, stream>>>(pooled, rhoW, rhob, clsW, clsb, out);
}
